// TorchTreeNNModel_61753039781888
// MI455X (gfx1250) — matrix operand order repaired and checked by reading, not compiled, not run on hardware
//
#include <hip/hip_runtime.h>
#include <cmath>

// Problem constants (match reference)
#define B_TOTAL 4096
#define SEQ_L   128
#define EDIM    128
#define OUT_N   3

// Tiling
#define BT      8                 // batch items per block
#define NWAVES  16
#define BLOCK   (NWAVES * 32)     // 512 threads, wave32
#define WPITCH  264               // W_tree LDS row pitch in f16 (256 + 8 pad -> 528B, bank-spread)
#define RPITCH  136               // reps LDS node-row pitch in f16 (128 + 8 pad -> 272B)

// LDS layout (dynamic, 277KB of the 320KB WGP LDS)
#define SMEM_W    (128 * WPITCH * 2)        // 67584 B : W_tree as f16
#define SMEM_BIAS (128 * 4)                 //   512 B : b_tree f32
#define SMEM_B0   (512 * RPITCH * 2)        // 139264 B: reps ping (level-1 out, 512 node rows)
#define SMEM_B1   (256 * RPITCH * 2)        //  69632 B: reps pong (256 node rows)
#define SMEM_TOTAL (SMEM_W + SMEM_BIAS + SMEM_B0 + SMEM_B1)

typedef __attribute__((ext_vector_type(16))) _Float16 v16h;
typedef __attribute__((ext_vector_type(8)))  _Float16 v8h;
typedef __attribute__((ext_vector_type(8)))  float    v8f;

union Frag16 { v16h v; v8h h[2]; };

// Fast tanh via TRANS ops (v_exp_f32 / v_rcp_f32 co-execute with XDL WMMA on CDNA5).
// tanh(x) = sign(x) * (1-e)/(1+e), e = exp2(-2*log2(e)*|x|); e->0 for large |x| (safe saturation).
__device__ __forceinline__ float fast_tanh(float x) {
  const float ax = fabsf(x);
  const float e  = __builtin_amdgcn_exp2f(-2.8853900818f * ax);
  const float r  = __builtin_amdgcn_rcpf(1.0f + e);
  const float t  = (1.0f - e) * r;
  return copysignf(t, x);
}

// B fragment (32x16 K-major) for f16 WMMA: lane n (0..15) holds W[ntile*16+n][K {0..7, 16..23}],
// lanes 16..31 hold K {8..15, 24..31}. Two 16B loads per lane from row-major W in LDS.
__device__ __forceinline__ v16h load_b_frag(const _Float16* __restrict__ Wlds,
                                            int ntile, int c, int lane) {
  const int n = ntile * 16 + (lane & 15);
  const int k = c * 32 + ((lane >> 4) << 3);
  const _Float16* p = Wlds + n * WPITCH + k;
  Frag16 f;
  f.h[0] = *(const v8h*)(p);
  f.h[1] = *(const v8h*)(p + 16);
  return f.v;
}

// A fragment (16x32) from LDS reps: A-row m (256 wide) = node rows 2m,2m+1 (128 each).
// 16-bit A layout: lanes 0..15 -> K {0..7, 16..23}; lanes 16..31 -> K {8..15, 24..31}.
__device__ __forceinline__ v16h load_a_frag_lds(const _Float16* __restrict__ buf,
                                                int mbase, int c, int lane) {
  const int m    = mbase + (lane & 15);
  const int half = (lane >> 4) & 1;
  const int h0   = c * 32 + half * 8;
  const int h1   = h0 + 16;
  const _Float16* p0 = buf + (2 * m + (h0 >> 7)) * RPITCH + (h0 & 127);
  const _Float16* p1 = buf + (2 * m + (h1 >> 7)) * RPITCH + (h1 & 127);
  Frag16 f;
  f.h[0] = *(const v8h*)p0;
  f.h[1] = *(const v8h*)p1;
  return f.v;
}

// Level-1 A fragment straight from the global embedding table (L2-resident gather):
// row m's 256-wide input = [emb[tokL] ; emb[tokR]], converted f32->f16 in registers.
__device__ __forceinline__ v16h gather_a_frag(const float* __restrict__ emb,
                                              int tokL, int tokR, int c, int lane) {
  const int half = (lane >> 4) & 1;
  const int h0   = c * 32 + half * 8;
  const int h1   = h0 + 16;
  const float* p0 = emb + (size_t)((h0 & 128) ? tokR : tokL) * EDIM + (h0 & 127);
  const float* p1 = emb + (size_t)((h1 & 128) ? tokR : tokL) * EDIM + (h1 & 127);
  float4 x0 = *(const float4*)(p0);
  float4 x1 = *(const float4*)(p0 + 4);
  float4 y0 = *(const float4*)(p1);
  float4 y1 = *(const float4*)(p1 + 4);
  v16h r;
  r[0]  = (_Float16)x0.x; r[1]  = (_Float16)x0.y; r[2]  = (_Float16)x0.z; r[3]  = (_Float16)x0.w;
  r[4]  = (_Float16)x1.x; r[5]  = (_Float16)x1.y; r[6]  = (_Float16)x1.z; r[7]  = (_Float16)x1.w;
  r[8]  = (_Float16)y0.x; r[9]  = (_Float16)y0.y; r[10] = (_Float16)y0.z; r[11] = (_Float16)y0.w;
  r[12] = (_Float16)y1.x; r[13] = (_Float16)y1.y; r[14] = (_Float16)y1.z; r[15] = (_Float16)y1.w;
  return r;
}

// C/D 16x16 f32 layout: VGPR r, lane L -> M = r + (L>=16 ? 8 : 0), N = L&15.
// Uniform fast path (full tile) avoids per-store exec-mask churn; guarded path only
// runs for the single padded tile at level 7 (Mout=8).
__device__ __forceinline__ void store_tile(_Float16* __restrict__ outb, const v8f* acc,
                                           const float* __restrict__ biasLds,
                                           int s, int Mout, int lane) {
  const int colBase = lane & 15;
  const int rowOff  = (lane >> 4) * 8;
  if (s * 16 + 16 <= Mout) {
    #pragma unroll
    for (int nt = 0; nt < 8; ++nt) {
      const int col  = nt * 16 + colBase;
      const float bv = biasLds[col];
      #pragma unroll
      for (int r = 0; r < 8; ++r) {
        const int m = s * 16 + rowOff + r;
        outb[m * RPITCH + col] = (_Float16)fast_tanh(acc[nt][r] + bv);
      }
    }
  } else {
    #pragma unroll
    for (int nt = 0; nt < 8; ++nt) {
      const int col  = nt * 16 + colBase;
      const float bv = biasLds[col];
      #pragma unroll
      for (int r = 0; r < 8; ++r) {
        const int m = s * 16 + rowOff + r;
        if (m < Mout) outb[m * RPITCH + col] = (_Float16)fast_tanh(acc[nt][r] + bv);
      }
    }
  }
}

__global__ __launch_bounds__(BLOCK)
void tree_nn_kernel(const int*   __restrict__ tokens,
                    const float* __restrict__ emb,
                    const float* __restrict__ Wt,     // (128, 256) row-major
                    const float* __restrict__ bt,     // (128)
                    const float* __restrict__ Wc,     // (3, 128) row-major
                    const float* __restrict__ bc,     // (3)
                    float*       __restrict__ out)    // (4096, 3)
{
  extern __shared__ char smem[];
  _Float16* Wlds    = (_Float16*)smem;
  float*    biasLds = (float*)(smem + SMEM_W);
  _Float16* buf0    = (_Float16*)(smem + SMEM_W + SMEM_BIAS);
  _Float16* buf1    = buf0 + 512 * RPITCH;

  const int tid  = threadIdx.x;
  const int lane = tid & 31;
  const int wave = tid >> 5;
  const int bb   = blockIdx.x * BT;   // first batch item of this block

  // ---- Stage W_tree (f32 -> f16) and bias into LDS ----
  {
    const int row = tid >> 2;         // 128 rows, 4 threads per row
    const int q   = tid & 3;
    const float* src = Wt + row * 256 + q * 64;
    _Float16*    dst = Wlds + row * WPITCH + q * 64;
    #pragma unroll
    for (int i = 0; i < 64; i += 4) {
      float4 v = *(const float4*)(src + i);
      dst[i + 0] = (_Float16)v.x; dst[i + 1] = (_Float16)v.y;
      dst[i + 2] = (_Float16)v.z; dst[i + 3] = (_Float16)v.w;
    }
  }
  if (tid < 128) biasLds[tid] = bt[tid];
  __syncthreads();

  // ---- Level 1: fused gather + GEMM. M = BT*64 = 512 rows -> 32 stripes over 16 waves ----
  for (int s = wave; s < 32; s += NWAVES) {
    v8f acc[8] = {};
    const int m  = s * 16 + (lane & 15);       // output row handled by this lane
    const int b  = m >> 6;
    const int j  = m & 63;
    const size_t tbase = (size_t)(bb + b) * SEQ_L + 2 * j;
    const int tokL = tokens[tbase];
    const int tokR = tokens[tbase + 1];
    #pragma unroll
    for (int c = 0; c < 8; ++c) {
      v16h a = gather_a_frag(emb, tokL, tokR, c, lane);
      #pragma unroll
      for (int nt = 0; nt < 8; ++nt) {
        v16h bfrag = load_b_frag(Wlds, nt, c, lane);
        acc[nt] = __builtin_amdgcn_wmma_f32_16x16x32_f16(
            false, a, false, bfrag, (short)0, acc[nt], false, false);
      }
    }
    store_tile(buf0, acc, biasLds, s, 512, lane);
  }
  __syncthreads();

  // ---- Levels 2..7: LDS ping-pong. Level L: Mout = BT * 64 >> (L-1) ----
  _Float16* inb  = buf0;
  _Float16* outb = buf1;
  int Mout = 256;
  for (int lev = 2; lev <= 7; ++lev) {
    const int Mtiles = (Mout + 15) >> 4;       // last level: 8 rows padded to 1 tile
    for (int s = wave; s < Mtiles; s += NWAVES) {
      v8f acc[8] = {};
      #pragma unroll
      for (int c = 0; c < 8; ++c) {
        v16h a = load_a_frag_lds(inb, s * 16, c, lane);
        #pragma unroll
        for (int nt = 0; nt < 8; ++nt) {
          v16h bfrag = load_b_frag(Wlds, nt, c, lane);
          acc[nt] = __builtin_amdgcn_wmma_f32_16x16x32_f16(
              false, a, false, bfrag, (short)0, acc[nt], false, false);
        }
      }
      store_tile(outb, acc, biasLds, s, Mout, lane);
    }
    __syncthreads();
    _Float16* t = inb; inb = outb; outb = t;
    Mout >>= 1;
  }
  // After the loop, roots (BT rows x 128, f16) live in `inb` (== buf0).

  // ---- Classifier: out[b, o] = roots[b] . W_cls[o] + b_cls[o]  (24 dot products) ----
  if (tid < BT * OUT_N) {
    const int b = tid / OUT_N;
    const int o = tid % OUT_N;
    const _Float16* rr = inb + b * RPITCH;
    const float*    wr = Wc + o * EDIM;
    float acc = bc[o];
    #pragma unroll 8
    for (int e = 0; e < EDIM; ++e) acc += (float)rr[e] * wr[e];
    out[(size_t)(bb + b) * OUT_N + o] = acc;
  }
}

extern "C" void kernel_launch(void* const* d_in, const int* in_sizes, int n_in,
                              void* d_out, int out_size, void* d_ws, size_t ws_size,
                              hipStream_t stream) {
  (void)in_sizes; (void)n_in; (void)d_ws; (void)ws_size; (void)out_size;
  const int*   tokens = (const int*)  d_in[0];
  const float* emb    = (const float*)d_in[1];
  const float* Wt     = (const float*)d_in[2];
  const float* bt     = (const float*)d_in[3];
  const float* Wc     = (const float*)d_in[4];
  const float* bc     = (const float*)d_in[5];
  float*       out    = (float*)d_out;

  dim3 grid(B_TOTAL / BT);   // 512 blocks, one per 8 sequences
  dim3 block(BLOCK);         // 512 threads = 16 wave32 waves
  tree_nn_kernel<<<grid, block, SMEM_TOTAL, stream>>>(tokens, emb, Wt, bt, Wc, bc, out);
}
